// GINEEncoderPP_14697378087540
// MI455X (gfx1250) — compile-verified, matrix operand order checked by reading, not run on hardware
//
#include <hip/hip_runtime.h>
#include <math.h>

// ---------------------------------------------------------------------------
// GINE encoder for gfx1250 (MI455X).  Dense per-node MLPs run on
// v_wmma_f32_16x16x32_f16 (f16 A/B staged in LDS, f32 accumulate); the edge
// gather/scatter phase uses global_atomic_add_f32 into an L2-resident
// accumulator (h+agg = 25.6 MB << 192 MB L2).
// ---------------------------------------------------------------------------

#define NNODES 50000
#define NEDGES 800000
#define NSLOPE 0.01f

typedef __attribute__((ext_vector_type(16))) _Float16 v16h;
typedef __attribute__((ext_vector_type(8)))  float    v8f;

// Build a WMMA 16-bit A/B fragment for this lane from a contiguous-K row in
// LDS.  Per the CDNA5 16-bit 16x32 layout: lane group g = lane/16 owns
// K = kbase+8g..+7 (regs 0-3) and K = kbase+16+8g..+7 (regs 4-7): two
// contiguous 8-half (16B) chunks -> ds_load_b128 friendly.
__device__ __forceinline__ v16h frag_row(const _Float16* __restrict__ row,
                                         int g, int kbase) {
  v16h r;
#pragma unroll
  for (int i = 0; i < 8; ++i) r[i]     = row[kbase + 8 * g + i];
#pragma unroll
  for (int i = 0; i < 8; ++i) r[8 + i] = row[kbase + 16 + 8 * g + i];
  return r;
}

__device__ __forceinline__ v8f wmma_f16(v16h a, v16h b, v8f c) {
  return __builtin_amdgcn_wmma_f32_16x16x32_f16(
      /*neg_a=*/false, a, /*neg_b=*/false, b,
      /*c_mod=*/(short)0, c, /*reuse_a=*/false, /*reuse_b=*/false);
}

__device__ __forceinline__ float lrelu(float v) {
  return v > 0.f ? v : NSLOPE * v;
}

// ---------------------------------------------------------------------------
// prep MLP: h = tanh(lrelu(x @ Wp_in + bp_in) @ Wp_h + bp_h)
// x: [N,16]  Wp_in: [16,64]  Wp_h: [64,64]  h: [N,64]
// block = 128 threads = 4 waves; each wave computes a 16x64 tile.
// ---------------------------------------------------------------------------
__global__ __launch_bounds__(128)
void k_prep(const float* __restrict__ x,
            const float* __restrict__ W0, const float* __restrict__ b0,
            const float* __restrict__ W1, const float* __restrict__ b1,
            float* __restrict__ h) {
  __shared__ _Float16 sW0T[64][32];   // Wp_in^T, K padded 16->32 with zeros
  __shared__ _Float16 sW1T[64][64];   // Wp_h^T
  __shared__ float sb0[64], sb1[64];
  __shared__ _Float16 sA[4][16][64];  // per-wave node tile (f16)

  const int tid  = threadIdx.x;
  const int lane = tid & 31;
  const int w    = tid >> 5;
  const int g    = lane >> 4;
  const int mn   = lane & 15;
  const int base = blockIdx.x * 64 + w * 16;

  for (int idx = tid; idx < 64 * 32; idx += 128) {
    int nn = idx >> 5, k = idx & 31;
    sW0T[nn][k] = (k < 16) ? (_Float16)W0[k * 64 + nn] : (_Float16)0.f;
  }
  for (int idx = tid; idx < 64 * 64; idx += 128) {
    int nn = idx >> 6, k = idx & 63;
    sW1T[nn][k] = (_Float16)W1[k * 64 + nn];
  }
  if (tid < 64) { sb0[tid] = b0[tid]; sb1[tid] = b1[tid]; }
  // stage x tile, zero-padding K 16..31
  for (int idx = lane; idx < 16 * 32; idx += 32) {
    int m = idx >> 5, k = idx & 31;
    int node = base + m; if (node >= NNODES) node = NNODES - 1;
    sA[w][m][k] = (k < 16) ? (_Float16)x[node * 16 + k] : (_Float16)0.f;
  }
  __syncthreads();

  // GEMM1: [16,32pad] x [32pad,64], then lrelu -> z tile back into sA
  v16h a = frag_row(&sA[w][mn][0], g, 0);
#pragma unroll
  for (int nt = 0; nt < 4; ++nt) {
    int n = nt * 16 + mn;
    v16h b = frag_row(&sW0T[n][0], g, 0);
    v8f c = {0.f, 0.f, 0.f, 0.f, 0.f, 0.f, 0.f, 0.f};
    c = wmma_f16(a, b, c);
#pragma unroll
    for (int r = 0; r < 8; ++r)
      sA[w][r + 8 * g][n] = (_Float16)lrelu(c[r] + sb0[n]);
  }
  __syncthreads();

  // GEMM2: [16,64] x [64,64], tanh -> h
  v16h a0 = frag_row(&sA[w][mn][0], g, 0);
  v16h a1 = frag_row(&sA[w][mn][0], g, 32);
#pragma unroll
  for (int nt = 0; nt < 4; ++nt) {
    int n = nt * 16 + mn;
    v16h b0v = frag_row(&sW1T[n][0], g, 0);
    v16h b1v = frag_row(&sW1T[n][0], g, 32);
    v8f c = {0.f, 0.f, 0.f, 0.f, 0.f, 0.f, 0.f, 0.f};
    c = wmma_f16(a0, b0v, c);
    c = wmma_f16(a1, b1v, c);
#pragma unroll
    for (int r = 0; r < 8; ++r) {
      int node = base + r + 8 * g;
      if (node < NNODES) h[node * 64 + n] = tanhf(c[r] + sb1[n]);
    }
  }
}

// ---------------------------------------------------------------------------
// zero agg[N*64] (float4 granularity)
// ---------------------------------------------------------------------------
__global__ __launch_bounds__(256)
void k_zero(float4* __restrict__ p, int n4) {
  int i = blockIdx.x * blockDim.x + threadIdx.x;
  if (i < n4) p[i] = make_float4(0.f, 0.f, 0.f, 0.f);
}

// ---------------------------------------------------------------------------
// edge phase: msg = relu(h[src] + w*We + be); agg[dst] += msg
// One thread per edge; gather + atomics stay in L2 (h/agg fully resident).
// ---------------------------------------------------------------------------
__global__ __launch_bounds__(256)
void k_scatter(const float* __restrict__ h,
               const int* __restrict__ src, const int* __restrict__ dst,
               const float* __restrict__ ew,
               const float* __restrict__ We, const float* __restrict__ be,
               float* __restrict__ agg) {
  int e = blockIdx.x * blockDim.x + threadIdx.x;
  if (e >= NEDGES) return;
  float wgt = ew[e];
  int s = src[e], d = dst[e];
  const float4* hs = (const float4*)(h + (size_t)s * 64);
  const float4* Wv = (const float4*)We;
  const float4* bv = (const float4*)be;
  float* ad = agg + (size_t)d * 64;
#pragma unroll
  for (int c = 0; c < 16; ++c) {
    float4 hv = hs[c], wv = Wv[c], bb = bv[c];
    float m0 = hv.x + wgt * wv.x + bb.x;
    float m1 = hv.y + wgt * wv.y + bb.y;
    float m2 = hv.z + wgt * wv.z + bb.z;
    float m3 = hv.w + wgt * wv.w + bb.w;
    m0 = m0 > 0.f ? m0 : 0.f;
    m1 = m1 > 0.f ? m1 : 0.f;
    m2 = m2 > 0.f ? m2 : 0.f;
    m3 = m3 > 0.f ? m3 : 0.f;
    atomicAdd(ad + 4 * c + 0, m0);
    atomicAdd(ad + 4 * c + 1, m1);
    atomicAdd(ad + 4 * c + 2, m2);
    atomicAdd(ad + 4 * c + 3, m3);
  }
}

// ---------------------------------------------------------------------------
// conv node MLP: z = agg + h; h = tanh(lrelu(z@W0+b0)@W1+b1)   (in place)
// ---------------------------------------------------------------------------
__global__ __launch_bounds__(128)
void k_conv_mlp(const float* __restrict__ hin, const float* __restrict__ agg,
                const float* __restrict__ W0, const float* __restrict__ b0,
                const float* __restrict__ W1, const float* __restrict__ b1,
                float* __restrict__ hout) {
  __shared__ _Float16 sW0T[64][64];
  __shared__ _Float16 sW1T[64][64];
  __shared__ float sb0[64], sb1[64];
  __shared__ _Float16 sA[4][16][64];

  const int tid  = threadIdx.x;
  const int lane = tid & 31;
  const int w    = tid >> 5;
  const int g    = lane >> 4;
  const int mn   = lane & 15;
  const int base = blockIdx.x * 64 + w * 16;

  for (int idx = tid; idx < 64 * 64; idx += 128) {
    int nn = idx >> 6, k = idx & 63;
    sW0T[nn][k] = (_Float16)W0[k * 64 + nn];
    sW1T[nn][k] = (_Float16)W1[k * 64 + nn];
  }
  if (tid < 64) { sb0[tid] = b0[tid]; sb1[tid] = b1[tid]; }
  // stage z = agg + h tile as f16
  for (int idx = lane; idx < 16 * 64; idx += 32) {
    int m = idx >> 6, k = idx & 63;
    int node = base + m; if (node >= NNODES) node = NNODES - 1;
    sA[w][m][k] = (_Float16)(hin[node * 64 + k] + agg[node * 64 + k]);
  }
  __syncthreads();

  v16h a0 = frag_row(&sA[w][mn][0], g, 0);
  v16h a1 = frag_row(&sA[w][mn][0], g, 32);
#pragma unroll
  for (int nt = 0; nt < 4; ++nt) {
    int n = nt * 16 + mn;
    v16h b0v = frag_row(&sW0T[n][0], g, 0);
    v16h b1v = frag_row(&sW0T[n][0], g, 32);
    v8f c = {0.f, 0.f, 0.f, 0.f, 0.f, 0.f, 0.f, 0.f};
    c = wmma_f16(a0, b0v, c);
    c = wmma_f16(a1, b1v, c);
#pragma unroll
    for (int r = 0; r < 8; ++r)
      sA[w][r + 8 * g][n] = (_Float16)lrelu(c[r] + sb0[n]);
  }
  __syncthreads();

  v16h z0 = frag_row(&sA[w][mn][0], g, 0);
  v16h z1 = frag_row(&sA[w][mn][0], g, 32);
#pragma unroll
  for (int nt = 0; nt < 4; ++nt) {
    int n = nt * 16 + mn;
    v16h b0v = frag_row(&sW1T[n][0], g, 0);
    v16h b1v = frag_row(&sW1T[n][0], g, 32);
    v8f c = {0.f, 0.f, 0.f, 0.f, 0.f, 0.f, 0.f, 0.f};
    c = wmma_f16(z0, b0v, c);
    c = wmma_f16(z1, b1v, c);
#pragma unroll
    for (int r = 0; r < 8; ++r) {
      int node = base + r + 8 * g;
      if (node < NNODES) hout[node * 64 + n] = tanhf(c[r] + sb1[n]);
    }
  }
}

// ---------------------------------------------------------------------------
// post MLP: out = tanh(lrelu(h@Wq0+bq0) @ Wq1 + bq1)   Wq1: [64,32]
// ---------------------------------------------------------------------------
__global__ __launch_bounds__(128)
void k_post(const float* __restrict__ hin,
            const float* __restrict__ W0, const float* __restrict__ b0,
            const float* __restrict__ W1, const float* __restrict__ b1,
            float* __restrict__ out) {
  __shared__ _Float16 sW0T[64][64];
  __shared__ _Float16 sW1T[32][64];
  __shared__ float sb0[64], sb1[32];
  __shared__ _Float16 sA[4][16][64];

  const int tid  = threadIdx.x;
  const int lane = tid & 31;
  const int w    = tid >> 5;
  const int g    = lane >> 4;
  const int mn   = lane & 15;
  const int base = blockIdx.x * 64 + w * 16;

  for (int idx = tid; idx < 64 * 64; idx += 128) {
    int nn = idx >> 6, k = idx & 63;
    sW0T[nn][k] = (_Float16)W0[k * 64 + nn];
  }
  for (int idx = tid; idx < 32 * 64; idx += 128) {
    int nn = idx >> 6, k = idx & 63;
    sW1T[nn][k] = (_Float16)W1[k * 32 + nn];
  }
  if (tid < 64) sb0[tid] = b0[tid];
  if (tid < 32) sb1[tid] = b1[tid];
  for (int idx = lane; idx < 16 * 64; idx += 32) {
    int m = idx >> 6, k = idx & 63;
    int node = base + m; if (node >= NNODES) node = NNODES - 1;
    sA[w][m][k] = (_Float16)hin[node * 64 + k];
  }
  __syncthreads();

  v16h a0 = frag_row(&sA[w][mn][0], g, 0);
  v16h a1 = frag_row(&sA[w][mn][0], g, 32);
#pragma unroll
  for (int nt = 0; nt < 4; ++nt) {
    int n = nt * 16 + mn;
    v16h b0v = frag_row(&sW0T[n][0], g, 0);
    v16h b1v = frag_row(&sW0T[n][0], g, 32);
    v8f c = {0.f, 0.f, 0.f, 0.f, 0.f, 0.f, 0.f, 0.f};
    c = wmma_f16(a0, b0v, c);
    c = wmma_f16(a1, b1v, c);
#pragma unroll
    for (int r = 0; r < 8; ++r)
      sA[w][r + 8 * g][n] = (_Float16)lrelu(c[r] + sb0[n]);
  }
  __syncthreads();

  v16h z0 = frag_row(&sA[w][mn][0], g, 0);
  v16h z1 = frag_row(&sA[w][mn][0], g, 32);
#pragma unroll
  for (int nt = 0; nt < 2; ++nt) {   // 32 output cols
    int n = nt * 16 + mn;
    v16h b0v = frag_row(&sW1T[n][0], g, 0);
    v16h b1v = frag_row(&sW1T[n][0], g, 32);
    v8f c = {0.f, 0.f, 0.f, 0.f, 0.f, 0.f, 0.f, 0.f};
    c = wmma_f16(z0, b0v, c);
    c = wmma_f16(z1, b1v, c);
#pragma unroll
    for (int r = 0; r < 8; ++r) {
      int node = base + r + 8 * g;
      if (node < NNODES) out[node * 32 + n] = tanhf(c[r] + sb1[n]);
    }
  }
}

// ---------------------------------------------------------------------------
extern "C" void kernel_launch(void* const* d_in, const int* in_sizes, int n_in,
                              void* d_out, int out_size, void* d_ws, size_t ws_size,
                              hipStream_t stream) {
  const float* x     = (const float*)d_in[0];
  const int*   ei    = (const int*)d_in[1];        // [2,E]: row0=src, row1=dst
  const float* ew    = (const float*)d_in[2];      // [E,1]
  const float* Wp_in = (const float*)d_in[3];
  const float* bp_in = (const float*)d_in[4];
  const float* Wp_h  = (const float*)d_in[5];
  const float* bp_h  = (const float*)d_in[6];
  const float* We    = (const float*)d_in[7];      // [3,1,64]
  const float* be    = (const float*)d_in[8];      // [3,64]
  const float* Wm0   = (const float*)d_in[9];      // [3,64,64]
  const float* bm0   = (const float*)d_in[10];
  const float* Wm1   = (const float*)d_in[11];
  const float* bm1   = (const float*)d_in[12];
  const float* Wq0   = (const float*)d_in[13];
  const float* bq0   = (const float*)d_in[14];
  const float* Wq1   = (const float*)d_in[15];     // [64,32]
  const float* bq1   = (const float*)d_in[16];

  float* h   = (float*)d_ws;                       // [N,64]
  float* agg = h + (size_t)NNODES * 64;            // [N,64]

  const int nblk = (NNODES + 63) / 64;             // 64 nodes / block (4 waves)

  k_prep<<<nblk, 128, 0, stream>>>(x, Wp_in, bp_in, Wp_h, bp_h, h);

  for (int k = 0; k < 3; ++k) {
    int n4 = NNODES * 16;                          // N*64 floats / 4
    k_zero<<<(n4 + 255) / 256, 256, 0, stream>>>((float4*)agg, n4);
    k_scatter<<<(NEDGES + 255) / 256, 256, 0, stream>>>(
        h, ei, ei + NEDGES, ew, We + k * 64, be + k * 64, agg);
    k_conv_mlp<<<nblk, 128, 0, stream>>>(
        h, agg, Wm0 + k * 4096, bm0 + k * 64, Wm1 + k * 4096, bm1 + k * 64, h);
  }

  k_post<<<nblk, 128, 0, stream>>>(h, Wq0, bq0, Wq1, bq1, (float*)d_out);
}
